// GNN_23794118820496
// MI455X (gfx1250) — matrix operand order repaired and checked by reading, not compiled, not run on hardware
//
#include <hip/hip_runtime.h>
#include <hip/hip_bf16.h>

// ---------------------------------------------------------------------------
// 3-layer GCN for MI455X (gfx1250, wave32, WMMA).
//   per layer: H = Xbf16 @ Wbf16 (v_wmma_f32_16x16x32_bf16, f32 accum)
//              Agg[dst] += H[src]  (f32 L2 atomics; H/Agg fit in 192MB L2)
//              X' = relu(Agg + b)  (fused bias+ReLU+bf16 convert)
// ---------------------------------------------------------------------------

typedef __attribute__((ext_vector_type(16))) __bf16 v16bf;
typedef __attribute__((ext_vector_type(8)))  float  v8f;

__device__ __forceinline__ unsigned short f32_bf16(float f) {
  unsigned u = __float_as_uint(f);
  u += 0x7FFFu + ((u >> 16) & 1u);           // round-to-nearest-even
  return (unsigned short)(u >> 16);
}

__global__ void k_f32_to_bf16(const float* __restrict__ src,
                              unsigned short* __restrict__ dst, int n) {
  int i = blockIdx.x * blockDim.x + threadIdx.x;
  if (i < n) dst[i] = f32_bf16(src[i]);
}

__global__ void k_zero(float* __restrict__ p, int n) {
  int i = blockIdx.x * blockDim.x + threadIdx.x;
  if (i < n) p[i] = 0.0f;
}

// Repack W [Fin, Fout] row-major f32 -> WMMA B-fragments (bf16).
// Fragment layout: [kt][nt][lane 0..31][j 0..15], where
//   col = nt*16 + (lane & 15),  k = kt*32 + (lane>>4)*16 + j
__global__ void k_repack_w(const float* __restrict__ W,
                           unsigned short* __restrict__ frag,
                           int Fin, int Fout) {
  int tid = blockIdx.x * blockDim.x + threadIdx.x;
  if (tid >= Fin * Fout) return;
  int j    = tid & 15;
  int lane = (tid >> 4) & 31;
  int rest = tid >> 9;
  int ntiles = Fout >> 4;
  int nt = rest % ntiles;
  int kt = rest / ntiles;
  int k   = kt * 32 + ((lane >> 4) << 4) + j;
  int col = nt * 16 + (lane & 15);
  frag[tid] = f32_bf16(W[(size_t)k * Fout + col]);
}

// GEMM: H[N, Fout] = A[N, FIN](bf16, row-major) x Bfrag.
// One wave -> 16(M) x 32(N) output strip; 8 waves/block cover 128 rows.
template <int FIN>
__global__ void k_gemm_wmma(const unsigned short* __restrict__ A,
                            const unsigned short* __restrict__ Bfrag,
                            float* __restrict__ H, int Nrows, int Fout) {
  const int lane  = threadIdx.x & 31;
  const int wave  = threadIdx.x >> 5;
  const int mTile = blockIdx.x * 8 + wave;
  const int m0    = mTile * 16;
  if (m0 >= Nrows) return;                      // wave-uniform guard
  const int nGroup = blockIdx.y;                // 32 output columns
  const int ntiles = Fout >> 4;

  // A-fragment addressing (contiguous K, matching Bfrag):
  //   lanes 0-15: row = lane,    K = 0..15
  //   lanes16-31: row = lane-16, K = 16..31
  const int row   = m0 + (lane & 15);
  const int kbase = (lane >> 4) * 16;
  const unsigned short* arow = A + (size_t)row * FIN;

  v8f acc0 = {};
  v8f acc1 = {};

#pragma unroll
  for (int kt = 0; kt < FIN / 32; ++kt) {
    union { v16bf v; uint4 q[2]; } a, b0, b1;
    a.q[0] = *(const uint4*)(arow + kt * 32 + kbase);        // K=kbase..+7
    a.q[1] = *(const uint4*)(arow + kt * 32 + kbase + 8);    // K=kbase+8..+15

    const unsigned short* bp0 =
        Bfrag + ((size_t)(kt * ntiles + nGroup * 2) * 512) + lane * 16;
    const unsigned short* bp1 =
        Bfrag + ((size_t)(kt * ntiles + nGroup * 2 + 1) * 512) + lane * 16;
    b0.q[0] = *(const uint4*)(bp0);
    b0.q[1] = *(const uint4*)(bp0 + 8);
    b1.q[0] = *(const uint4*)(bp1);
    b1.q[1] = *(const uint4*)(bp1 + 8);

    acc0 = __builtin_amdgcn_wmma_f32_16x16x32_bf16(
        false, a.v, false, b0.v, (short)0, acc0, false, false);
    acc1 = __builtin_amdgcn_wmma_f32_16x16x32_bf16(
        false, a.v, false, b1.v, (short)0, acc1, false, false);
  }

  // C/D layout: VGPR r -> row m0 + r + 8*(lane>>4), col = lane&15
  const int n0    = nGroup * 32 + (lane & 15);
  const int rbase = m0 + ((lane >> 4) << 3);
#pragma unroll
  for (int r = 0; r < 8; ++r) {
    float* hr = H + (size_t)(rbase + r) * Fout;
    hr[n0]      = acc0[r];
    hr[n0 + 16] = acc1[r];
  }
}

// Edge scatter: 64 threads per edge, each owns a float4 feature chunk.
// Agg[dst] += H[src]. f32 atomics resolve in L2 (both buffers L2-resident).
// All 32 lanes of a wave share one edge -> index loads coalesce to one
// broadcast request per wave.
__global__ void k_scatter(const int* __restrict__ ei,
                          const float* __restrict__ H,
                          float* __restrict__ Agg, int E) {
  int tid  = blockIdx.x * blockDim.x + threadIdx.x;
  int edge = tid >> 6;
  if (edge >= E) return;
  int c   = (tid & 63) * 4;
  int src = ei[edge];
  int dst = ei[(size_t)E + edge];
  float4 v = *(const float4*)(H + (size_t)src * 256 + c);
  float* p = Agg + (size_t)dst * 256 + c;
  unsafeAtomicAdd(p + 0, v.x);
  unsafeAtomicAdd(p + 1, v.y);
  unsafeAtomicAdd(p + 2, v.z);
  unsafeAtomicAdd(p + 3, v.w);
}

// Fused bias + ReLU; emits f32 (final layer) and/or bf16 (next layer input).
__global__ void k_epilogue(const float* __restrict__ Agg,
                           const float* __restrict__ bias,
                           float* __restrict__ out32,
                           unsigned short* __restrict__ out16, int total) {
  int i = blockIdx.x * blockDim.x + threadIdx.x;
  if (i >= total) return;
  float v = Agg[i] + bias[i & 255];
  v = v > 0.0f ? v : 0.0f;
  if (out32) out32[i] = v;
  if (out16) out16[i] = f32_bf16(v);
}

extern "C" void kernel_launch(void* const* d_in, const int* in_sizes, int n_in,
                              void* d_out, int out_size, void* d_ws,
                              size_t ws_size, hipStream_t stream) {
  const float* x  = (const float*)d_in[0];
  const int*   ei = (const int*)d_in[1];   // edge_index [2,E] as int32
  const float* W1 = (const float*)d_in[2];
  const float* b1 = (const float*)d_in[3];
  const float* W2 = (const float*)d_in[4];
  const float* b2 = (const float*)d_in[5];
  const float* W3 = (const float*)d_in[6];
  const float* b3 = (const float*)d_in[7];

  const int N = in_sizes[0] / 128;   // 50000 nodes
  const int E = in_sizes[1] / 2;     // 800000 edges

  // Workspace layout (~128 MB)
  char* ws = (char*)d_ws;
  unsigned short* Xb = (unsigned short*)ws;                       // N*256 bf16
  unsigned short* Wf = (unsigned short*)(ws + (size_t)N * 512);   // 128 KB
  float* H   = (float*)(ws + (size_t)N * 512 + 131072);           // N*256 f32
  float* Agg = (float*)(ws + (size_t)N * 512 + 131072 + (size_t)N * 1024);

  const int nTot     = N * 256;
  const int ewBlocks = (E * 64 + 255) / 256;
  const dim3 gGemm(((N + 15) / 16 + 7) / 8, 256 / 32);  // M-tiles/8 x Fout/32

  // ---- layer 1 ----
  k_f32_to_bf16<<<(N * 128 + 255) / 256, 256, 0, stream>>>(x, Xb, N * 128);
  k_repack_w<<<(128 * 256 + 255) / 256, 256, 0, stream>>>(W1, Wf, 128, 256);
  k_gemm_wmma<128><<<gGemm, 256, 0, stream>>>(Xb, Wf, H, N, 256);
  k_zero<<<(nTot + 255) / 256, 256, 0, stream>>>(Agg, nTot);
  k_scatter<<<ewBlocks, 256, 0, stream>>>(ei, H, Agg, E);
  k_epilogue<<<(nTot + 255) / 256, 256, 0, stream>>>(Agg, b1, nullptr, Xb, nTot);

  // ---- layer 2 ----
  k_repack_w<<<(256 * 256 + 255) / 256, 256, 0, stream>>>(W2, Wf, 256, 256);
  k_gemm_wmma<256><<<gGemm, 256, 0, stream>>>(Xb, Wf, H, N, 256);
  k_zero<<<(nTot + 255) / 256, 256, 0, stream>>>(Agg, nTot);
  k_scatter<<<ewBlocks, 256, 0, stream>>>(ei, H, Agg, E);
  k_epilogue<<<(nTot + 255) / 256, 256, 0, stream>>>(Agg, b2, nullptr, Xb, nTot);

  // ---- layer 3 ----
  k_repack_w<<<(256 * 256 + 255) / 256, 256, 0, stream>>>(W3, Wf, 256, 256);
  k_gemm_wmma<256><<<gGemm, 256, 0, stream>>>(Xb, Wf, H, N, 256);
  k_zero<<<(nTot + 255) / 256, 256, 0, stream>>>(Agg, nTot);
  k_scatter<<<ewBlocks, 256, 0, stream>>>(ei, H, Agg, E);
  k_epilogue<<<(nTot + 255) / 256, 256, 0, stream>>>(Agg, b3, (float*)d_out,
                                                     nullptr, nTot);
}
